// NBEATSBlock_26663156974243
// MI455X (gfx1250) — matrix operand order repaired and checked by reading, not compiled, not run on hardware
//
#include <hip/hip_runtime.h>

#define Bc 4096
#define Dc 1024
#define Hc 2048
#define Ec 8
#define Tc 1280
#define LN_EPS 1e-5f

typedef __attribute__((ext_vector_type(16))) __bf16 v16bf;
typedef __attribute__((ext_vector_type(8)))  __bf16 v8bf;
typedef __attribute__((ext_vector_type(8)))  float  v8f;

// ---------------------------------------------------------------------------
// Fragment loaders matching CDNA5 16-bit WMMA VGPR layouts (ISA 7.12.2).
// A 16x32 (MxK): lane m = lane&15; lanes>=16 take K-offset +8; elems 0..7 ->
// K = ka..ka+7, elems 8..15 -> K = 16+ka..16+ka+7.
// B 32x16 (KxN) from pre-transposed Bt[N][K]: lane n = lane&15; lanes>=16 take
// K-offset +8; elems 0..7 -> K = kb..kb+7, elems 8..15 -> K = 16+kb..16+kb+7.
// ---------------------------------------------------------------------------
__device__ inline v16bf load_a_frag(const __bf16* __restrict__ p) {
  v8bf lo = *(const v8bf*)p;
  v8bf hi = *(const v8bf*)(p + 16);
  return __builtin_shufflevector(lo, hi, 0,1,2,3,4,5,6,7,8,9,10,11,12,13,14,15);
}

__device__ inline v16bf load_b_frag(const __bf16* __restrict__ p) {
  v8bf lo = *(const v8bf*)p;
  v8bf hi = *(const v8bf*)(p + 16);
  return __builtin_shufflevector(lo, hi, 0,1,2,3,4,5,6,7,8,9,10,11,12,13,14,15);
}

// ---------------------------------------------------------------------------
// Routed bf16 WMMA GEMM over one expert's compacted row list.
//   C[slot, n] = epilogue( A[row(slot), :] * Bt[n, :]^T + bias[n] )
// Block: 256 threads = 8 waves as 2(M) x 4(N); wave tile 32x64 (2x4 frags);
// block tile 64 x 256. Grid.x is worst-case (Bc/64); blocks beyond the
// expert's row count exit early (count read from memory -> graph-safe).
// K-loop is manually double-buffered: next step's 6 fragments are issued
// before the current step's 8 WMMAs so loads stay in flight across the MACs.
// GATHER: A rows indirected through rows_e (first GEMM of the chain).
// MODE 0: store bf16 to compact slot     MODE 1: relu + store bf16
// MODE 2: scale by routed weight, scatter-accumulate into split fp32 output
//         (plain +=: expert launches are stream-serialized, unique writer)
// ---------------------------------------------------------------------------
template<int MODE, bool GATHER>
__global__ __launch_bounds__(256)
void gemm_wmma_bf16(const __bf16* __restrict__ A, const __bf16* __restrict__ Bt,
                    const float* __restrict__ bias, __bf16* __restrict__ Cout,
                    int Kdim, int N,
                    const int* __restrict__ cntp, const int* __restrict__ rows_e,
                    const float* __restrict__ wrow_e, float* __restrict__ outF) {
  const int cnt = cntp[0];
  if ((int)blockIdx.x * 64 >= cnt) return;   // routed early-exit

  const int lane = threadIdx.x & 31;
  const int wid  = threadIdx.x >> 5;
  const int wm   = wid >> 2;        // 0..1
  const int wn   = wid & 3;         // 0..3
  const int row0 = blockIdx.x * 64  + wm * 32;
  const int col0 = blockIdx.y * 256 + wn * 64;

  const int lm = lane & 15;
  const int ka = (lane >> 4) << 3;    // A frag K sub-offset (0 or 8)
  const int kb = (lane >> 4) << 3;    // B frag K sub-offset (0 or 8)

  // A row resolution (gather through routing table on first GEMM)
  int slotA0 = row0 + lm;
  int slotA1 = row0 + 16 + lm;
  int rA0, rA1;
  if (GATHER) {
    rA0 = rows_e[slotA0 < cnt ? slotA0 : cnt - 1];
    rA1 = rows_e[slotA1 < cnt ? slotA1 : cnt - 1];
  } else {
    rA0 = slotA0;
    rA1 = slotA1;
  }

  const __bf16* a0p = A + (size_t)rA0 * Kdim + ka;
  const __bf16* a1p = A + (size_t)rA1 * Kdim + ka;
  const __bf16* bp0 = Bt + (size_t)(col0 +  0 + lm) * Kdim + kb;
  const __bf16* bp1 = Bt + (size_t)(col0 + 16 + lm) * Kdim + kb;
  const __bf16* bp2 = Bt + (size_t)(col0 + 32 + lm) * Kdim + kb;
  const __bf16* bp3 = Bt + (size_t)(col0 + 48 + lm) * Kdim + kb;

  v8f acc[2][4];
#pragma unroll
  for (int fm = 0; fm < 2; ++fm)
#pragma unroll
    for (int fn = 0; fn < 4; ++fn)
      acc[fm][fn] = {};

  // prologue: fragments for k = 0
  v16bf a0 = load_a_frag(a0p);
  v16bf a1 = load_a_frag(a1p);
  v16bf b0 = load_b_frag(bp0);
  v16bf b1 = load_b_frag(bp1);
  v16bf b2 = load_b_frag(bp2);
  v16bf b3 = load_b_frag(bp3);

  for (int k0 = 0; k0 < Kdim - 32; k0 += 32) {
    const int kn = k0 + 32;
    // issue next generation's loads before consuming the current one
    v16bf na0 = load_a_frag(a0p + kn);
    v16bf na1 = load_a_frag(a1p + kn);
    v16bf nb0 = load_b_frag(bp0 + kn);
    v16bf nb1 = load_b_frag(bp1 + kn);
    v16bf nb2 = load_b_frag(bp2 + kn);
    v16bf nb3 = load_b_frag(bp3 + kn);

    acc[0][0] = __builtin_amdgcn_wmma_f32_16x16x32_bf16(false, a0, false, b0, (short)0, acc[0][0], false, false);
    acc[1][0] = __builtin_amdgcn_wmma_f32_16x16x32_bf16(false, a1, false, b0, (short)0, acc[1][0], false, false);
    acc[0][1] = __builtin_amdgcn_wmma_f32_16x16x32_bf16(false, a0, false, b1, (short)0, acc[0][1], false, false);
    acc[1][1] = __builtin_amdgcn_wmma_f32_16x16x32_bf16(false, a1, false, b1, (short)0, acc[1][1], false, false);
    acc[0][2] = __builtin_amdgcn_wmma_f32_16x16x32_bf16(false, a0, false, b2, (short)0, acc[0][2], false, false);
    acc[1][2] = __builtin_amdgcn_wmma_f32_16x16x32_bf16(false, a1, false, b2, (short)0, acc[1][2], false, false);
    acc[0][3] = __builtin_amdgcn_wmma_f32_16x16x32_bf16(false, a0, false, b3, (short)0, acc[0][3], false, false);
    acc[1][3] = __builtin_amdgcn_wmma_f32_16x16x32_bf16(false, a1, false, b3, (short)0, acc[1][3], false, false);

    a0 = na0; a1 = na1; b0 = nb0; b1 = nb1; b2 = nb2; b3 = nb3;
  }

  // epilogue K-step
  acc[0][0] = __builtin_amdgcn_wmma_f32_16x16x32_bf16(false, a0, false, b0, (short)0, acc[0][0], false, false);
  acc[1][0] = __builtin_amdgcn_wmma_f32_16x16x32_bf16(false, a1, false, b0, (short)0, acc[1][0], false, false);
  acc[0][1] = __builtin_amdgcn_wmma_f32_16x16x32_bf16(false, a0, false, b1, (short)0, acc[0][1], false, false);
  acc[1][1] = __builtin_amdgcn_wmma_f32_16x16x32_bf16(false, a1, false, b1, (short)0, acc[1][1], false, false);
  acc[0][2] = __builtin_amdgcn_wmma_f32_16x16x32_bf16(false, a0, false, b2, (short)0, acc[0][2], false, false);
  acc[1][2] = __builtin_amdgcn_wmma_f32_16x16x32_bf16(false, a1, false, b2, (short)0, acc[1][2], false, false);
  acc[0][3] = __builtin_amdgcn_wmma_f32_16x16x32_bf16(false, a0, false, b3, (short)0, acc[0][3], false, false);
  acc[1][3] = __builtin_amdgcn_wmma_f32_16x16x32_bf16(false, a1, false, b3, (short)0, acc[1][3], false, false);

  // C/D layout: lane n = lane&15; VGPR r holds M = r + (lane>=16 ? 8 : 0)
  const int nloc  = lane & 15;
  const int rbase = (lane >> 4) << 3;
#pragma unroll
  for (int fm = 0; fm < 2; ++fm) {
#pragma unroll
    for (int fn = 0; fn < 4; ++fn) {
      const int gcol = col0 + fn * 16 + nloc;
      const float bv = bias[gcol];
#pragma unroll
      for (int r = 0; r < 8; ++r) {
        const int slot = row0 + fm * 16 + rbase + r;
        float v = acc[fm][fn][r] + bv;
        if (MODE == 1) v = fmaxf(v, 0.0f);
        if (MODE == 0 || MODE == 1) {
          // padded slots (slot >= cnt) hold garbage; masked at final scatter
          Cout[(size_t)slot * N + gcol] = (__bf16)v;
        } else {
          if (slot < cnt) {
            const int   orow = rows_e[slot];
            const float w    = wrow_e[slot];
            v *= w;
            if (gcol < Dc) {
              float* p = outF + (size_t)orow * Dc + gcol;
              *p = *p + v;
            } else {
              float* p = outF + (size_t)Bc * Dc + (size_t)orow * (Tc - Dc) + (gcol - Dc);
              *p = *p + v;
            }
          }
        }
      }
    }
  }
}

// ---------------------------------------------------------------------------
// LayerNorm + gate logits + softmax + top-2 -> per-row expert weights
// One 256-thread block per row; D = 1024 -> 4 elems/thread.
// ---------------------------------------------------------------------------
__global__ __launch_bounds__(256)
void gate_kernel(const float* __restrict__ x, const float* __restrict__ gamma,
                 const float* __restrict__ beta, const float* __restrict__ Wg,
                 const float* __restrict__ bg, float* __restrict__ wexp) {
  __shared__ float red[256];
  __shared__ float s_stat[2];
  __shared__ float s_logit[Ec];
  const int b = blockIdx.x;
  const int t = threadIdx.x;
  const float* xr = x + (size_t)b * Dc;

  float lx[4];
  float s = 0.0f;
#pragma unroll
  for (int i = 0; i < 4; ++i) { lx[i] = xr[t + i * 256]; s += lx[i]; }

  red[t] = s; __syncthreads();
  for (int o = 128; o > 0; o >>= 1) { if (t < o) red[t] += red[t + o]; __syncthreads(); }
  if (t == 0) s_stat[0] = red[0] * (1.0f / Dc);
  __syncthreads();
  const float mu = s_stat[0];
  __syncthreads();

  float sq = 0.0f;
#pragma unroll
  for (int i = 0; i < 4; ++i) { float d = lx[i] - mu; sq += d * d; }
  red[t] = sq; __syncthreads();
  for (int o = 128; o > 0; o >>= 1) { if (t < o) red[t] += red[t + o]; __syncthreads(); }
  if (t == 0) s_stat[1] = rsqrtf(red[0] * (1.0f / Dc) + LN_EPS);
  __syncthreads();
  const float rstd = s_stat[1];
  __syncthreads();

  float pl[Ec];
#pragma unroll
  for (int e = 0; e < Ec; ++e) pl[e] = 0.0f;
#pragma unroll
  for (int i = 0; i < 4; ++i) {
    const int idx = t + i * 256;
    const float xn = (lx[i] - mu) * rstd * gamma[idx] + beta[idx];
    const float* wr = Wg + (size_t)idx * Ec;
#pragma unroll
    for (int e = 0; e < Ec; ++e) pl[e] += xn * wr[e];
  }
  for (int e = 0; e < Ec; ++e) {
    red[t] = pl[e]; __syncthreads();
    for (int o = 128; o > 0; o >>= 1) { if (t < o) red[t] += red[t + o]; __syncthreads(); }
    if (t == 0) s_logit[e] = red[0] + bg[e];
    __syncthreads();
  }

  if (t == 0) {
    float l[Ec];
#pragma unroll
    for (int e = 0; e < Ec; ++e) l[e] = s_logit[e];
    int i1 = 0;
    for (int e = 1; e < Ec; ++e) if (l[e] > l[i1]) i1 = e;      // first max
    int i2 = (i1 == 0) ? 1 : 0;
    for (int e = 0; e < Ec; ++e) if (e != i1 && l[e] > l[i2]) i2 = e;
    // softmax normalization cancels in the top-2 ratio
    const float e2 = expf(l[i2] - l[i1]);
    const float w1 = 1.0f / (1.0f + e2);
    const float w2 = e2 / (1.0f + e2);
    float* wr = wexp + (size_t)b * Ec;
#pragma unroll
    for (int e = 0; e < Ec; ++e) wr[e] = 0.0f;
    wr[i1] = w1;
    wr[i2] = w2;
  }
}

// ---------------------------------------------------------------------------
// Routing: compact per-expert row lists. Slot order is nondeterministic but
// each row's result is slot-invariant, so output stays deterministic.
// ---------------------------------------------------------------------------
__global__ __launch_bounds__(256)
void route_kernel(const float* __restrict__ wexp, int* __restrict__ cnt,
                  int* __restrict__ rows, float* __restrict__ wrow) {
  const int b = blockIdx.x * 256 + threadIdx.x;
  if (b >= Bc) return;
  const float* wr = wexp + (size_t)b * Ec;
#pragma unroll
  for (int e = 0; e < Ec; ++e) {
    const float w = wr[e];
    if (w > 0.0f) {
      const int s = atomicAdd(&cnt[e], 1);
      rows[(size_t)e * Bc + s] = b;
      wrow[(size_t)e * Bc + s] = w;
    }
  }
}

// ---------------------------------------------------------------------------
__global__ __launch_bounds__(256)
void cvt_bf16_kernel(const float* __restrict__ in, __bf16* __restrict__ out, size_t n) {
  size_t i = (size_t)blockIdx.x * blockDim.x + threadIdx.x;
  if (i < n) out[i] = (__bf16)in[i];
}

// fp32 [R][C] -> bf16 transposed [C][R], batched over experts via blockIdx.z
__global__ __launch_bounds__(256)
void tconv_kernel(const float* __restrict__ in, __bf16* __restrict__ out, int R, int C) {
  __shared__ float tile[32][33];
  const size_t base = (size_t)blockIdx.z * R * C;
  const int c0 = blockIdx.x * 32;
  const int r0 = blockIdx.y * 32;
  const int tx = threadIdx.x, ty = threadIdx.y;
#pragma unroll
  for (int i = 0; i < 32; i += 8)
    tile[ty + i][tx] = in[base + (size_t)(r0 + ty + i) * C + (c0 + tx)];
  __syncthreads();
#pragma unroll
  for (int i = 0; i < 32; i += 8)
    out[base + (size_t)(c0 + ty + i) * R + (r0 + tx)] = (__bf16)tile[tx][ty + i];
}

__global__ __launch_bounds__(256)
void zero_b32_kernel(unsigned* __restrict__ p, size_t n) {
  size_t i = (size_t)blockIdx.x * blockDim.x + threadIdx.x;
  if (i < n) p[i] = 0u;
}

// ---------------------------------------------------------------------------
extern "C" void kernel_launch(void* const* d_in, const int* in_sizes, int n_in,
                              void* d_out, int out_size, void* d_ws, size_t ws_size,
                              hipStream_t stream) {
  const float* x     = (const float*)d_in[0];
  const float* gamma = (const float*)d_in[1];
  const float* beta  = (const float*)d_in[2];
  const float* Wg    = (const float*)d_in[3];
  const float* bg    = (const float*)d_in[4];
  const float* W_in  = (const float*)d_in[5];
  const float* b_in  = (const float*)d_in[6];
  const float* W1    = (const float*)d_in[7];
  const float* b1    = (const float*)d_in[8];
  const float* W2    = (const float*)d_in[9];
  const float* b2    = (const float*)d_in[10];
  const float* W_out = (const float*)d_in[11];
  const float* b_out = (const float*)d_in[12];
  float* out = (float*)d_out;

  // workspace carve-up (256B aligned)
  char* ws = (char*)d_ws;
  size_t off = 0;
  auto carve = [&](size_t bytes) -> void* {
    void* p = ws + off;
    off += (bytes + 255) & ~(size_t)255;
    return p;
  };
  float*  wexp  = (float*) carve((size_t)Bc * Ec * 4);
  int*    cnt   = (int*)   carve((size_t)Ec * 4);
  int*    rows  = (int*)   carve((size_t)Ec * Bc * 4);
  float*  wrow  = (float*) carve((size_t)Ec * Bc * 4);
  __bf16* xb    = (__bf16*)carve((size_t)Bc * Dc * 2);
  __bf16* wtin  = (__bf16*)carve((size_t)Ec * Hc * Dc * 2);   // [e][H][D]
  __bf16* wt1   = (__bf16*)carve((size_t)Ec * Hc * Hc * 2);   // [e][H][H]
  __bf16* wt2   = (__bf16*)carve((size_t)Ec * Hc * Hc * 2);
  __bf16* wtout = (__bf16*)carve((size_t)Ec * Tc * Hc * 2);   // [e][T][H]
  __bf16* h1    = (__bf16*)carve((size_t)Bc * Hc * 2);
  __bf16* h2    = (__bf16*)carve((size_t)Bc * Hc * 2);

  // 1) zero output accumulator + routing counters
  {
    size_t n = (size_t)Bc * Tc;
    zero_b32_kernel<<<dim3((unsigned)((n + 255) / 256)), dim3(256), 0, stream>>>((unsigned*)out, n);
    zero_b32_kernel<<<dim3(1), dim3(256), 0, stream>>>((unsigned*)cnt, (size_t)Ec);
  }

  // 2) gate weights + routing tables
  gate_kernel<<<dim3(Bc), dim3(256), 0, stream>>>(x, gamma, beta, Wg, bg, wexp);
  route_kernel<<<dim3(Bc / 256), dim3(256), 0, stream>>>(wexp, cnt, rows, wrow);

  // 3) pack activations + weights to bf16 (weights transposed to [N][K])
  {
    size_t n = (size_t)Bc * Dc;
    cvt_bf16_kernel<<<dim3((unsigned)((n + 255) / 256)), dim3(256), 0, stream>>>(x, xb, n);
  }
  tconv_kernel<<<dim3(Hc / 32, Dc / 32, Ec), dim3(32, 8), 0, stream>>>(W_in,  wtin,  Dc, Hc);
  tconv_kernel<<<dim3(Hc / 32, Hc / 32, Ec), dim3(32, 8), 0, stream>>>(W1,    wt1,   Hc, Hc);
  tconv_kernel<<<dim3(Hc / 32, Hc / 32, Ec), dim3(32, 8), 0, stream>>>(W2,    wt2,   Hc, Hc);
  tconv_kernel<<<dim3(Tc / 32, Hc / 32, Ec), dim3(32, 8), 0, stream>>>(W_out, wtout, Hc, Tc);

  // 4) per-expert routed WMMA GEMM chain (serialized on stream; h1/h2 ping-pong)
  const dim3 gridH(Bc / 64, Hc / 256);   // worst-case rows; blocks exit on cnt
  const dim3 gridT(Bc / 64, Tc / 256);
  for (int e = 0; e < Ec; ++e) {
    const int*   cnt_e  = cnt + e;
    const int*   rows_e = rows + (size_t)e * Bc;
    const float* wrow_e = wrow + (size_t)e * Bc;
    gemm_wmma_bf16<0, true ><<<gridH, dim3(256), 0, stream>>>(
        xb, wtin + (size_t)e * Hc * Dc, b_in + (size_t)e * Hc, h1,
        Dc, Hc, cnt_e, rows_e, wrow_e, nullptr);
    gemm_wmma_bf16<1, false><<<gridH, dim3(256), 0, stream>>>(
        h1, wt1 + (size_t)e * Hc * Hc, b1 + (size_t)e * Hc, h2,
        Hc, Hc, cnt_e, rows_e, wrow_e, nullptr);
    gemm_wmma_bf16<1, false><<<gridH, dim3(256), 0, stream>>>(
        h2, wt2 + (size_t)e * Hc * Hc, b2 + (size_t)e * Hc, h1,
        Hc, Hc, cnt_e, rows_e, wrow_e, nullptr);
    gemm_wmma_bf16<2, false><<<gridT, dim3(256), 0, stream>>>(
        h1, wtout + (size_t)e * Tc * Hc, b_out + (size_t)e * Tc, nullptr,
        Hc, Tc, cnt_e, rows_e, wrow_e, out);
  }
}
